// GNNRoutingModel_1425929142866
// MI455X (gfx1250) — matrix operand order repaired and checked by reading, not compiled, not run on hardware
//
#include <hip/hip_runtime.h>
#include <stdint.h>

// ---------------------------------------------------------------------------
// Model dims (from reference)
// ---------------------------------------------------------------------------
constexpr int T_STEPS = 8;
constexpr int N_NODES = 4096;
constexpr int N_EDGES = 131072;
constexpr int IN_DIM  = 64;
constexpr int HID     = 128;
constexpr int OUTD    = 2;

// ---------------------------------------------------------------------------
// WMMA vector types (CDNA5, wave32)
// ---------------------------------------------------------------------------
typedef float v8f __attribute__((ext_vector_type(8)));
typedef __bf16 bf16x16 __attribute__((ext_vector_type(16)));

union FragBF {
    uint4   q[2];   // two 16B chunks loaded from LDS
    bf16x16 v;      // the 16-lane-element bf16 WMMA operand
};

__device__ __forceinline__ unsigned short f32_to_bf16(float f) {
    union { float f; unsigned int u; } cv; cv.f = f;
    unsigned int u = cv.u;
    u += 0x7FFFu + ((u >> 16) & 1u);   // round-to-nearest-even
    return (unsigned short)(u >> 16);
}

// ---------------------------------------------------------------------------
// Weight-folding prep kernels (exploit hc = concat(h,h))
// ---------------------------------------------------------------------------
__global__ __launch_bounds__(256) void prep_fc1c(const float* __restrict__ fc1w,
                                                 float* __restrict__ fc1c) {
    int idx = blockIdx.x * 256 + threadIdx.x;        // HID*HID
    int k = idx >> 7, f = idx & 127;
    fc1c[idx] = fc1w[k * HID + f] + fc1w[(k + HID) * HID + f];
}

__global__ __launch_bounds__(256) void prep_fchc(const float* __restrict__ fchw,
                                                 unsigned short* __restrict__ fchc) {
    int idx = blockIdx.x * 256 + threadIdx.x;        // HID*N_NODES
    int k = idx >> 12, n = idx & (N_NODES - 1);
    fchc[idx] = f32_to_bf16(fchw[k * N_NODES + n] + fchw[(k + HID) * N_NODES + n]);
}

// ---------------------------------------------------------------------------
// GCN normalization: deg = 1 (self loop) + scatter(ew, dst); dinv = rsqrt(deg)
// ---------------------------------------------------------------------------
__global__ __launch_bounds__(256) void deg_init(float* __restrict__ deg) {
    deg[blockIdx.x * 256 + threadIdx.x] = 1.0f;
}

__global__ __launch_bounds__(256) void deg_accum(const int* __restrict__ dst,
                                                 const float* __restrict__ ew,
                                                 float* __restrict__ deg) {
    int e = blockIdx.x * 256 + threadIdx.x;
    atomicAdd(&deg[dst[e]], ew[e]);
}

__global__ __launch_bounds__(256) void dinv_kernel(const float* __restrict__ deg,
                                                   float* __restrict__ dinv) {
    int i = blockIdx.x * 256 + threadIdx.x;
    float d = deg[i];
    dinv[i] = d > 0.0f ? rsqrtf(fmaxf(d, 1e-12f)) : 0.0f;
}

// ---------------------------------------------------------------------------
// Small f32 GEMV-style transform: Y[i][f] = dot(X[i][:K], W[:,f]),  W = [K][HID]
// ---------------------------------------------------------------------------
__global__ __launch_bounds__(HID) void linear_kernel(const float* __restrict__ X,
                                                     const float* __restrict__ W,
                                                     float* __restrict__ Y, int K) {
    __shared__ float xs[HID];
    int i = blockIdx.x, f = threadIdx.x;
    if (f < K) xs[f] = X[(size_t)i * K + f];
    __syncthreads();
    float s = 0.0f;
    for (int k = 0; k < K; ++k) s += xs[k] * W[k * HID + f];
    Y[(size_t)i * HID + f] = s;
}

// Self-loop contribution: agg[i] = dinv[i]^2 * hW[i]
__global__ __launch_bounds__(HID) void agg_init(const float* __restrict__ hW,
                                                const float* __restrict__ dinv,
                                                float* __restrict__ agg) {
    int i = blockIdx.x, f = threadIdx.x;
    float di = dinv[i];
    agg[(size_t)i * HID + f] = di * di * hW[(size_t)i * HID + f];
}

// Edge scatter: agg[dst] += dinv[src]*ew*dinv[dst] * hW[src]   (2 edges / block)
__global__ __launch_bounds__(256) void scatter_kernel(const int* __restrict__ src,
                                                      const int* __restrict__ dst,
                                                      const float* __restrict__ ew,
                                                      const float* __restrict__ dinv,
                                                      const float* __restrict__ hW,
                                                      float* __restrict__ agg) {
    int e = blockIdx.x * 2 + (threadIdx.x >> 7);
    int f = threadIdx.x & 127;
    int s = src[e], d = dst[e];
    float nrm = dinv[s] * ew[e] * dinv[d];
    atomicAdd(&agg[(size_t)d * HID + f], nrm * hW[(size_t)s * HID + f]);
}

// h = relu(agg + b); optionally also emit bf16 copy for the WMMA GEMM
__global__ __launch_bounds__(HID) void bias_relu(const float* __restrict__ agg,
                                                 const float* __restrict__ b,
                                                 float* __restrict__ h,
                                                 unsigned short* __restrict__ hbf) {
    int i = blockIdx.x, f = threadIdx.x;
    float v = fmaxf(agg[(size_t)i * HID + f] + b[f], 0.0f);
    h[(size_t)i * HID + f] = v;
    if (hbf) hbf[(size_t)i * HID + f] = f32_to_bf16(v);
}

// ---------------------------------------------------------------------------
// link = relu(h @ fc1c + fc1_b) @ fc2_w + fc2_b      (one node per block)
// ---------------------------------------------------------------------------
__global__ __launch_bounds__(HID) void link_kernel(const float* __restrict__ h,
                                                   const float* __restrict__ fc1c,
                                                   const float* __restrict__ fc1b,
                                                   const float* __restrict__ fc2w,
                                                   const float* __restrict__ fc2b,
                                                   float* __restrict__ link) {
    __shared__ float hs[HID];
    __shared__ float zs[HID];
    int i = blockIdx.x, f = threadIdx.x;
    hs[f] = h[(size_t)i * HID + f];
    __syncthreads();
    float s = fc1b[f];
    for (int k = 0; k < HID; ++k) s += hs[k] * fc1c[k * HID + f];
    zs[f] = fmaxf(s, 0.0f);
    __syncthreads();
    if (f < OUTD) {
        float o = fc2b[f];
        for (int k = 0; k < HID; ++k) o += zs[k] * fc2w[k * OUTD + f];
        link[(size_t)i * OUTD + f] = o;
    }
}

// ---------------------------------------------------------------------------
// hop = h(bf16) @ fchc(bf16) + fch_b   — bf16 WMMA GEMM, 128x64 tile per block
//   grid = (N/64, N/128), 256 threads = 8 waves, 2x2 16x16 tiles per wave
//   A tile staged with CDNA5 async global->LDS copies (ASYNCcnt path)
// ---------------------------------------------------------------------------
constexpr int LSTR = HID + 8;   // padded LDS row stride (elements) -> 272B rows

__global__ __launch_bounds__(256) void hop_wmma(const unsigned short* __restrict__ hbf,
                                                const unsigned short* __restrict__ wbf,
                                                const float* __restrict__ bias,
                                                float* __restrict__ out) {
    __shared__ __align__(16) unsigned short As[128 * LSTR];  // [row][k]
    __shared__ __align__(16) unsigned short Bs[64 * LSTR];   // [col][k] (transposed)

    const int bn  = blockIdx.x;   // 64-col tile
    const int bm  = blockIdx.y;   // 128-row tile
    const int tid = threadIdx.x;

    // Stage A: 128 rows x 128 k of bf16 via async copy: each lane moves 16B
    // straight global->LDS (no VGPR round-trip), tracked on ASYNCcnt.
    #pragma unroll
    for (int it = 0; it < 8; ++it) {
        int id  = it * 256 + tid;
        int row = id >> 4;
        int kc  = (id & 15) * 8;
        unsigned lds_addr = (unsigned)(uintptr_t)(As + row * LSTR + kc);
        unsigned long long gaddr =
            (unsigned long long)(uintptr_t)(hbf + (size_t)(bm * 128 + row) * HID + kc);
        asm volatile("global_load_async_to_lds_b128 %0, %1, off"
                     :: "v"(lds_addr), "v"(gaddr) : "memory");
    }
    // Stage B with transpose: global [k][n] (coalesced b128) -> LDS [n][k]
    // (byte-for-byte async copy cannot transpose, so this stays on the VGPR path)
    #pragma unroll
    for (int it = 0; it < 4; ++it) {
        int id = it * 256 + tid;
        int k  = id >> 3;
        int nc = (id & 7) * 8;
        uint4 v = *(const uint4*)(wbf + (size_t)k * N_NODES + bn * 64 + nc);
        const unsigned short* e = (const unsigned short*)&v;
        #pragma unroll
        for (int j = 0; j < 8; ++j) Bs[(nc + j) * LSTR + k] = e[j];
    }
    asm volatile("s_wait_asynccnt 0x0" ::: "memory");   // our async copies landed
    __syncthreads();                                    // everyone's staging landed

    const int wave = tid >> 5;
    const int lane = tid & 31;
    const int wm = wave >> 1;          // 32-row band within 128
    const int wn = wave & 1;           // 32-col band within 64
    const int lr = lane & 15;
    const int lh = lane >> 4;

    v8f acc[2][2] = {};
    FragBF a[2], b[2];

    #pragma unroll
    for (int ks = 0; ks < 4; ++ks) {
        // A fragment (16x32 bf16): per ISA layout, lane half lh selects the
        // interleaved K-split: [kb..kb+7] in v0-3 and [kb+16..kb+23] in v4-7
        #pragma unroll
        for (int mt = 0; mt < 2; ++mt) {
            int row = wm * 32 + mt * 16 + lr;
            int ka  = ks * 32 + lh * 8;
            a[mt].q[0] = *(const uint4*)(As + row * LSTR + ka);
            a[mt].q[1] = *(const uint4*)(As + row * LSTR + ka + 16);
        }
        // B fragment (32x16 bf16): same interleaved K-split as A: [kb..kb+7] and [kb+16..kb+23]
        #pragma unroll
        for (int nt = 0; nt < 2; ++nt) {
            int col = wn * 32 + nt * 16 + lr;
            int kb  = ks * 32 + lh * 8;
            b[nt].q[0] = *(const uint4*)(Bs + col * LSTR + kb);
            b[nt].q[1] = *(const uint4*)(Bs + col * LSTR + kb + 16);
        }
        #pragma unroll
        for (int mt = 0; mt < 2; ++mt)
            #pragma unroll
            for (int nt = 0; nt < 2; ++nt)
                acc[mt][nt] = __builtin_amdgcn_wmma_f32_16x16x32_bf16(
                    false, a[mt].v, false, b[nt].v,
                    (short)0, acc[mt][nt], false, false);
    }

    // C/D layout: VGPR j -> M = j + 8*(lane>=16), N = lane%16
    #pragma unroll
    for (int nt = 0; nt < 2; ++nt) {
        int gcol = bn * 64 + wn * 32 + nt * 16 + lr;
        float bv = bias[gcol];
        #pragma unroll
        for (int mt = 0; mt < 2; ++mt) {
            int rbase = bm * 128 + wm * 32 + mt * 16 + lh * 8;
            #pragma unroll
            for (int j = 0; j < 8; ++j)
                out[(size_t)(rbase + j) * N_NODES + gcol] = acc[mt][nt][j] + bv;
        }
    }
}

// ---------------------------------------------------------------------------
// Host driver
// ---------------------------------------------------------------------------
extern "C" void kernel_launch(void* const* d_in, const int* in_sizes, int n_in,
                              void* d_out, int out_size, void* d_ws, size_t ws_size,
                              hipStream_t stream) {
    const float* x          = (const float*)d_in[0];
    const int*   edge_index = (const int*)d_in[1];
    const float* edge_attr  = (const float*)d_in[2];
    const float* W1    = (const float*)d_in[3];
    const float* b1    = (const float*)d_in[4];
    const float* W2    = (const float*)d_in[5];
    const float* b2    = (const float*)d_in[6];
    const float* fc1_w = (const float*)d_in[7];
    const float* fc1_b = (const float*)d_in[8];
    const float* fc2_w = (const float*)d_in[9];
    const float* fc2_b = (const float*)d_in[10];
    const float* fch_w = (const float*)d_in[11];
    const float* fch_b = (const float*)d_in[12];

    float* link_out = (float*)d_out;                                    // [T][N][2]
    float* hop_out  = (float*)d_out + (size_t)T_STEPS * N_NODES * OUTD; // [T][N][N]

    char*  ws  = (char*)d_ws;
    size_t off = 0;
    auto alloc = [&](size_t bytes) -> void* {
        void* p = ws + off;
        off = (off + bytes + 255) & ~(size_t)255;
        return p;
    };
    float*          fc1c = (float*)alloc((size_t)HID * HID * 4);
    unsigned short* fchc = (unsigned short*)alloc((size_t)HID * N_NODES * 2);
    float*          deg  = (float*)alloc((size_t)N_NODES * 4);
    float*          dinv = (float*)alloc((size_t)N_NODES * 4);
    float*          hW   = (float*)alloc((size_t)N_NODES * HID * 4);
    float*          agg  = (float*)alloc((size_t)N_NODES * HID * 4);
    float*          h1   = (float*)alloc((size_t)N_NODES * HID * 4);
    float*          h2   = (float*)alloc((size_t)N_NODES * HID * 4);
    unsigned short* h2bf = (unsigned short*)alloc((size_t)N_NODES * HID * 2);

    prep_fc1c<<<(HID * HID) / 256, 256, 0, stream>>>(fc1_w, fc1c);
    prep_fchc<<<(HID * N_NODES) / 256, 256, 0, stream>>>(fch_w, fchc);

    for (int t = 0; t < T_STEPS; ++t) {
        const float* xt  = x + (size_t)t * N_NODES * IN_DIM;
        const int*   src = edge_index + (size_t)t * 2 * N_EDGES;
        const int*   dst = src + N_EDGES;
        const float* ew  = edge_attr + (size_t)t * N_EDGES;
        float* link_t = link_out + (size_t)t * N_NODES * OUTD;
        float* hop_t  = hop_out + (size_t)t * N_NODES * N_NODES;

        // symmetric normalization
        deg_init<<<N_NODES / 256, 256, 0, stream>>>(deg);
        deg_accum<<<N_EDGES / 256, 256, 0, stream>>>(dst, ew, deg);
        dinv_kernel<<<N_NODES / 256, 256, 0, stream>>>(deg, dinv);

        // conv1
        linear_kernel<<<N_NODES, HID, 0, stream>>>(xt, W1, hW, IN_DIM);
        agg_init<<<N_NODES, HID, 0, stream>>>(hW, dinv, agg);
        scatter_kernel<<<N_EDGES / 2, 256, 0, stream>>>(src, dst, ew, dinv, hW, agg);
        bias_relu<<<N_NODES, HID, 0, stream>>>(agg, b1, h1, (unsigned short*)nullptr);

        // conv2 (also emit bf16 activations for the WMMA GEMM)
        linear_kernel<<<N_NODES, HID, 0, stream>>>(h1, W2, hW, HID);
        agg_init<<<N_NODES, HID, 0, stream>>>(hW, dinv, agg);
        scatter_kernel<<<N_EDGES / 2, 256, 0, stream>>>(src, dst, ew, dinv, hW, agg);
        bias_relu<<<N_NODES, HID, 0, stream>>>(agg, b2, h2, h2bf);

        // link head (tiny)
        link_kernel<<<N_NODES, HID, 0, stream>>>(h2, fc1c, fc1_b, fc2_w, fc2_b, link_t);

        // hop head: the 34-GFLOP / 537-MB-store GEMM on the WMMA path
        dim3 hg(N_NODES / 64, N_NODES / 128);
        hop_wmma<<<hg, 256, 0, stream>>>(h2bf, fchc, fch_b, hop_t);
    }
}
